// InformerEncoder_29283087024332
// MI455X (gfx1250) — matrix operand order repaired and checked by reading, not compiled, not run on hardware
//
#include <hip/hip_runtime.h>
#include <hip/hip_bf16.h>
#include <math.h>

typedef __attribute__((ext_vector_type(16))) __bf16 v16bf;
typedef __attribute__((ext_vector_type(8)))  float  v8f;

union FragU { uint4 u[2]; v16bf v; };

__device__ inline unsigned short f2bf(float f) {
    unsigned int u = __float_as_uint(f);
    unsigned int r = u + 0x7FFFu + ((u >> 16) & 1u);
    if ((u & 0x7F800000u) == 0x7F800000u) r = u;   // inf/nan passthrough
    return (unsigned short)(r >> 16);
}

__device__ inline float gelu_f(float x) {
    return 0.5f * x * (1.0f + erff(x * 0.7071067811865476f));
}

// ---------------- elementwise conversion ----------------
__global__ void tobf_kernel(const float* __restrict__ src, unsigned short* __restrict__ dst, size_t n) {
    size_t i = (size_t)blockIdx.x * blockDim.x + threadIdx.x;
    if (i < n) dst[i] = f2bf(src[i]);
}

// pack conv weights (D,D,3) -> bf16 [3][D][D]
__global__ void pack_convw_kernel(const float* __restrict__ w, unsigned short* __restrict__ dst, int D) {
    int idx = blockIdx.x * blockDim.x + threadIdx.x;
    int n = 3 * D * D;
    if (idx >= n) return;
    int i = idx % D;
    int o = (idx / D) % D;
    int k = idx / (D * D);
    dst[idx] = f2bf(w[((size_t)o * D + i) * 3 + k]);
}

// ---------------- embedding conv + positional encoding ----------------
__global__ void embed_kernel(const float* __restrict__ x_enc, const float* __restrict__ emb_w,
                             float* __restrict__ out, int B, int L, int D, int Cc) {
    int idx = blockIdx.x * blockDim.x + threadIdx.x;
    if (idx >= B * L * D) return;
    int d = idx % D;
    int l = (idx / D) % L;
    int b = idx / (D * L);
    float s = 0.f;
    for (int k = 0; k < 3; ++k) {
        int ls = l + k - 1; ls = (ls + L) % L;
        const float* xp = x_enc + ((size_t)b * L + ls) * Cc;
        for (int c = 0; c < Cc; ++c)
            s += xp[c] * emb_w[((size_t)d * Cc + c) * 3 + k];
    }
    int p = d >> 1;
    float div = expf((float)(2 * p) * (-9.210340371976184f / (float)D)); // -ln(10000)/D
    float ang = (float)l * div;
    s += (d & 1) ? cosf(ang) : sinf(ang);
    out[idx] = s;
}

// ---------------- WMMA GEMM: C[M,N] = A[M,K] * W[N,K]^T + bias (+optional GELU) ----
// Fragment layouts per CDNA5 ISA:
//  A 16x32 bf16: lane r=lane&15, g=lane>>4; elems 0-7 = K[g*8 .. +7], elems 8-15 = K[16+g*8 .. +7]
//  B 32x16 bf16: lane n=lane&15, g=lane>>4; elems 0-7 = K[g*8 .. +7], elems 8-15 = K[16+g*8 .. +7] of column n (= W row n)
//  C/D f32 16x16: n = lane&15, m = j + 8*(lane>>4)
__device__ inline v16bf load_fragA(const unsigned short* base, int stride, int row, int k0, int lane) {
    int r = row + (lane & 15);
    int g = lane >> 4;
    const unsigned short* p = base + (size_t)r * stride + k0 + g * 8;
    FragU f;
    f.u[0] = *reinterpret_cast<const uint4*>(p);
    f.u[1] = *reinterpret_cast<const uint4*>(p + 16);
    return f.v;
}
__device__ inline v16bf load_fragB(const unsigned short* base, int stride, int col, int k0, int lane) {
    int n = col + (lane & 15);
    int g = lane >> 4;
    const unsigned short* p = base + (size_t)n * stride + k0 + g * 8;
    FragU f;
    f.u[0] = *reinterpret_cast<const uint4*>(p);
    f.u[1] = *reinterpret_cast<const uint4*>(p + 16);
    return f.v;
}

// one wave computes a 32x32 output macro-tile (2 M-tiles x 2 N-tiles),
// software-pipelined (double-buffered) over K so loads overlap WMMAs.
// Live regs at WMMA block: 32 acc + 32 cur frags + 32 next frags -> no spills.
__global__ void gemm_bf16_kernel(const unsigned short* __restrict__ A,
                                 const unsigned short* __restrict__ W,
                                 const float* __restrict__ bias,
                                 float* __restrict__ C,
                                 int M, int N, int K, int act) {
    int wave = (blockIdx.x * blockDim.x + threadIdx.x) >> 5;
    int lane = threadIdx.x & 31;
    int ntn = N >> 5;                       // 32-wide strips
    int total = (M >> 5) * ntn;             // 32-row strips
    if (wave >= total) return;              // wave-uniform: EXEC stays all-ones
    int tm = wave / ntn, tn = wave % ntn;
    int row0 = tm << 5, col0 = tn << 5;

    v8f acc00 = (v8f){0.f, 0.f, 0.f, 0.f, 0.f, 0.f, 0.f, 0.f};
    v8f acc01 = acc00, acc10 = acc00, acc11 = acc00;

    // prologue: fragments for k0 = 0
    v16bf a0 = load_fragA(A, K, row0, 0, lane);
    v16bf a1 = load_fragA(A, K, row0 + 16, 0, lane);
    v16bf b0 = load_fragB(W, K, col0, 0, lane);
    v16bf b1 = load_fragB(W, K, col0 + 16, 0, lane);

    for (int k0 = 0; k0 < K; k0 += 32) {
        int kn = k0 + 32;
        v16bf an0 = a0, an1 = a1, bn0 = b0, bn1 = b1;
        if (kn < K) {                       // issue next k-step loads before compute
            an0 = load_fragA(A, K, row0, kn, lane);
            an1 = load_fragA(A, K, row0 + 16, kn, lane);
            bn0 = load_fragB(W, K, col0, kn, lane);
            bn1 = load_fragB(W, K, col0 + 16, kn, lane);
            __builtin_prefetch(A + (size_t)(row0 + (lane & 15)) * K + kn + 32, 0, 1);
            __builtin_prefetch(W + (size_t)(col0 + (lane & 15)) * K + kn + 32, 0, 1);
        }
        acc00 = __builtin_amdgcn_wmma_f32_16x16x32_bf16(false, a0, false, b0, (short)0, acc00, false, false);
        acc10 = __builtin_amdgcn_wmma_f32_16x16x32_bf16(false, a1, false, b0, (short)0, acc10, false, false);
        acc01 = __builtin_amdgcn_wmma_f32_16x16x32_bf16(false, a0, false, b1, (short)0, acc01, false, false);
        acc11 = __builtin_amdgcn_wmma_f32_16x16x32_bf16(false, a1, false, b1, (short)0, acc11, false, false);
        a0 = an0; a1 = an1; b0 = bn0; b1 = bn1;
    }

    int g = lane >> 4;
    v8f accs[2][2] = {{acc00, acc01}, {acc10, acc11}};
    #pragma unroll
    for (int r = 0; r < 2; ++r) {
        #pragma unroll
        for (int c = 0; c < 2; ++c) {
            int n = col0 + c * 16 + (lane & 15);
            float bv = bias ? bias[n] : 0.f;
            #pragma unroll
            for (int j = 0; j < 8; ++j) {
                int m = row0 + r * 16 + j + (g << 3);
                float v = accs[r][c][j] + bv;
                if (act == 1) v = gelu_f(v);
                C[(size_t)m * N + n] = v;
            }
        }
    }
}

// circular conv1d (kernel 3) over channels as 3 accumulated WMMA GEMM passes,
// 16x64 output strip per wave, double-buffered over K within each tap
__global__ void conv3_gemm_kernel(const unsigned short* __restrict__ Xbf,
                                  const unsigned short* __restrict__ Wk3,
                                  float* __restrict__ C, int Bn, int L, int D) {
    int wave = (blockIdx.x * blockDim.x + threadIdx.x) >> 5;
    int lane = threadIdx.x & 31;
    int ntn = D >> 6;
    int total = ((Bn * L) >> 4) * ntn;
    if (wave >= total) return;
    int tm = wave / ntn, tn = wave % ntn;
    int row0 = tm << 4, col0 = tn << 6;
    int b = row0 / L, l0 = row0 % L;
    int g = lane >> 4, lr = lane & 15;
    v8f acc[4];
    #pragma unroll
    for (int c = 0; c < 4; ++c) acc[c] = (v8f){0.f, 0.f, 0.f, 0.f, 0.f, 0.f, 0.f, 0.f};
    for (int k = 0; k < 3; ++k) {
        int lsrc = l0 + lr + k - 1;
        lsrc = (lsrc + L) % L;                           // circular rows, per lane
        const unsigned short* arow = Xbf + ((size_t)b * L + lsrc) * D + g * 8;
        const unsigned short* wk = Wk3 + (size_t)k * D * D;

        FragU fa, fb[4];
        fa.u[0] = *reinterpret_cast<const uint4*>(arow);
        fa.u[1] = *reinterpret_cast<const uint4*>(arow + 16);
        #pragma unroll
        for (int c = 0; c < 4; ++c) {
            const unsigned short* brow = wk + (size_t)(col0 + c * 16 + lr) * D + g * 8;
            fb[c].u[0] = *reinterpret_cast<const uint4*>(brow);
            fb[c].u[1] = *reinterpret_cast<const uint4*>(brow + 16);
        }
        for (int k0 = 0; k0 < D; k0 += 32) {
            int kn = k0 + 32;
            FragU na = fa, nb0 = fb[0], nb1 = fb[1], nb2 = fb[2], nb3 = fb[3];
            if (kn < D) {
                na.u[0] = *reinterpret_cast<const uint4*>(arow + kn);
                na.u[1] = *reinterpret_cast<const uint4*>(arow + kn + 16);
                const unsigned short* br0 = wk + (size_t)(col0 + 0 * 16 + lr) * D + g * 8 + kn;
                const unsigned short* br1 = wk + (size_t)(col0 + 1 * 16 + lr) * D + g * 8 + kn;
                const unsigned short* br2 = wk + (size_t)(col0 + 2 * 16 + lr) * D + g * 8 + kn;
                const unsigned short* br3 = wk + (size_t)(col0 + 3 * 16 + lr) * D + g * 8 + kn;
                nb0.u[0] = *reinterpret_cast<const uint4*>(br0);
                nb0.u[1] = *reinterpret_cast<const uint4*>(br0 + 16);
                nb1.u[0] = *reinterpret_cast<const uint4*>(br1);
                nb1.u[1] = *reinterpret_cast<const uint4*>(br1 + 16);
                nb2.u[0] = *reinterpret_cast<const uint4*>(br2);
                nb2.u[1] = *reinterpret_cast<const uint4*>(br2 + 16);
                nb3.u[0] = *reinterpret_cast<const uint4*>(br3);
                nb3.u[1] = *reinterpret_cast<const uint4*>(br3 + 16);
            }
            acc[0] = __builtin_amdgcn_wmma_f32_16x16x32_bf16(false, fa.v, false, fb[0].v, (short)0, acc[0], false, false);
            acc[1] = __builtin_amdgcn_wmma_f32_16x16x32_bf16(false, fa.v, false, fb[1].v, (short)0, acc[1], false, false);
            acc[2] = __builtin_amdgcn_wmma_f32_16x16x32_bf16(false, fa.v, false, fb[2].v, (short)0, acc[2], false, false);
            acc[3] = __builtin_amdgcn_wmma_f32_16x16x32_bf16(false, fa.v, false, fb[3].v, (short)0, acc[3], false, false);
            fa = na; fb[0] = nb0; fb[1] = nb1; fb[2] = nb2; fb[3] = nb3;
        }
    }
    #pragma unroll
    for (int c = 0; c < 4; ++c) {
        int n = col0 + c * 16 + lr;
        #pragma unroll
        for (int j = 0; j < 8; ++j) {
            int m = row0 + j + (g << 3);
            C[(size_t)m * D + n] = acc[c][j];
        }
    }
}

// ---------------- layernorm (optionally fused residual), one wave per row, D=512 ---
__global__ void layernorm_kernel(const float* __restrict__ x, const float* __restrict__ res,
                                 const float* __restrict__ g, const float* __restrict__ bt,
                                 float* __restrict__ out, int D) {
    int row = blockIdx.x;
    int lane = threadIdx.x;          // blockDim = 32
    const float* xr = x + (size_t)row * D;
    const float* rr = res ? res + (size_t)row * D : nullptr;
    float vals[16];
    int nper = D / 32;
    float s = 0.f;
    for (int i = 0; i < nper; ++i) {
        float v = xr[lane + i * 32];
        if (rr) v += rr[lane + i * 32];
        vals[i] = v; s += v;
    }
    for (int m = 16; m >= 1; m >>= 1) s += __shfl_xor(s, m, 32);
    float mean = s / (float)D;
    float q = 0.f;
    for (int i = 0; i < nper; ++i) { float d0 = vals[i] - mean; q += d0 * d0; }
    for (int m = 16; m >= 1; m >>= 1) q += __shfl_xor(q, m, 32);
    float inv = rsqrtf(q / (float)D + 1e-5f);
    for (int i = 0; i < nper; ++i) {
        int d = lane + i * 32;
        out[(size_t)row * D + d] = (vals[i] - mean) * inv * g[d] + bt[d];
    }
}

// ---------------- ProbSparse attention (fp32 scalar path) ----------------
// M[b,h,l] = max_j q.k_sample - sum_j/L ; one wave per (b,h,l)
__global__ void qk_sample_M_kernel(const float* __restrict__ Q, const float* __restrict__ Km,
                                   float* __restrict__ M, int B, int Hh, int L, int U) {
    int wid = (blockIdx.x * blockDim.x + threadIdx.x) >> 5;
    int lane = threadIdx.x & 31;
    int total = B * Hh * L;
    if (wid >= total) return;
    int l = wid % L;
    int h = (wid / L) % Hh;
    int b = wid / (L * Hh);
    const float* qrow = Q + ((size_t)b * L + l) * (Hh * 64) + h * 64;
    float vmax = -3.4e38f, vsum = 0.f;
    for (int j = lane; j < U; j += 32) {
        unsigned int hs = (unsigned int)l * 2654435761u ^ ((unsigned int)j * 0x9E3779B9u);
        hs ^= hs >> 16; hs *= 0x85EBCA6Bu; hs ^= hs >> 13;
        int ls = (int)(hs % (unsigned)L);
        const float* krow = Km + ((size_t)b * L + ls) * (Hh * 64) + h * 64;
        float dot = 0.f;
        for (int e = 0; e < 64; ++e) dot += qrow[e] * krow[e];
        vmax = fmaxf(vmax, dot);
        vsum += dot;
    }
    for (int m = 16; m >= 1; m >>= 1) {
        vmax = fmaxf(vmax, __shfl_xor(vmax, m, 32));
        vsum += __shfl_xor(vsum, m, 32);
    }
    if (lane == 0) M[wid] = vmax - vsum / (float)L;
}

// top-u selection per (b,h); one wave per block
__global__ void topk_kernel(const float* __restrict__ M, int* __restrict__ Mtop,
                            int L, int u) {
    __shared__ float s[2048];
    int bh = blockIdx.x;
    int lane = threadIdx.x;          // 32
    const float* m = M + (size_t)bh * L;
    for (int i = lane; i < L; i += 32) s[i] = m[i];
    __syncthreads();
    for (int t = 0; t < u; ++t) {
        float bv = -3.4e38f; int bi = 0x7FFFFFFF;
        for (int i = lane; i < L; i += 32) {
            float v = s[i];
            if (v > bv || (v == bv && i < bi)) { bv = v; bi = i; }
        }
        for (int msk = 16; msk >= 1; msk >>= 1) {
            float ov = __shfl_xor(bv, msk, 32);
            int   oi = __shfl_xor(bi, msk, 32);
            if (ov > bv || (ov == bv && oi < bi)) { bv = ov; bi = oi; }
        }
        if (lane == 0) { Mtop[bh * u + t] = bi; s[bi] = -3.4e38f; }
        __syncthreads();
    }
}

// softmax(QK^T/8) @ V for selected queries; one block per (b,h,t)
__global__ void attn_upd_kernel(const float* __restrict__ Q, const float* __restrict__ Km,
                                const float* __restrict__ V, const int* __restrict__ Mtop,
                                float* __restrict__ upd, int B, int Hh, int L, int u) {
    __shared__ float sc[2048];
    __shared__ float red[256];
    __shared__ float qs[64];
    __shared__ float part[256];
    int blk = blockIdx.x;
    int t = blk % u;
    int h = (blk / u) % Hh;
    int b = blk / (u * Hh);
    int tid = threadIdx.x;           // 256
    int bh = b * Hh + h;
    int lq = Mtop[bh * u + t];
    if (tid < 64) qs[tid] = Q[((size_t)b * L + lq) * (Hh * 64) + h * 64 + tid];
    __syncthreads();
    float lmax = -3.4e38f;
    for (int l = tid; l < L; l += 256) {
        const float* krow = Km + ((size_t)b * L + l) * (Hh * 64) + h * 64;
        float dot = 0.f;
        for (int e = 0; e < 64; ++e) dot += qs[e] * krow[e];
        dot *= 0.125f;               // 1/sqrt(64)
        sc[l] = dot;
        lmax = fmaxf(lmax, dot);
    }
    red[tid] = lmax; __syncthreads();
    for (int s2 = 128; s2 > 0; s2 >>= 1) { if (tid < s2) red[tid] = fmaxf(red[tid], red[tid + s2]); __syncthreads(); }
    float mx = red[0]; __syncthreads();
    float lsum = 0.f;
    for (int l = tid; l < L; l += 256) { float e2 = expf(sc[l] - mx); sc[l] = e2; lsum += e2; }
    red[tid] = lsum; __syncthreads();
    for (int s2 = 128; s2 > 0; s2 >>= 1) { if (tid < s2) red[tid] += red[tid + s2]; __syncthreads(); }
    float inv = 1.f / red[0];
    int e = tid & 63, seg = tid >> 6;
    int chunk = L >> 2;
    float acc = 0.f;
    for (int l = seg * chunk; l < (seg + 1) * chunk; ++l)
        acc += sc[l] * V[((size_t)b * L + l) * (Hh * 64) + h * 64 + e];
    part[seg * 64 + e] = acc; __syncthreads();
    if (tid < 64)
        upd[((size_t)bh * u + t) * 64 + tid] =
            (part[tid] + part[64 + tid] + part[128 + tid] + part[192 + tid]) * inv;
}

__global__ void vmean_kernel(const float* __restrict__ V, float* __restrict__ vm,
                             int Hh, int L) {
    int bh = blockIdx.x;
    int e = threadIdx.x;             // 64
    int b = bh / Hh, h = bh % Hh;
    float s = 0.f;
    for (int l = 0; l < L; ++l) s += V[((size_t)b * L + l) * (Hh * 64) + h * 64 + e];
    vm[bh * 64 + e] = s / (float)L;
}

__global__ void ctx_fill_kernel(const float* __restrict__ vm, float* __restrict__ attn,
                                int B, int Hh, int L) {
    int D = Hh * 64;
    int idx = blockIdx.x * blockDim.x + threadIdx.x;
    if (idx >= B * L * D) return;
    int d = idx % D;
    int b = idx / (D * L);
    int h = d >> 6, e = d & 63;
    attn[idx] = vm[(b * Hh + h) * 64 + e];
}

__global__ void ctx_scatter_kernel(const float* __restrict__ upd, const int* __restrict__ Mtop,
                                   float* __restrict__ attn, int Hh, int L, int u) {
    int blk = blockIdx.x;
    int e = threadIdx.x;             // 64
    int t = blk % u;
    int h = (blk / u) % Hh;
    int b = blk / (u * Hh);
    int bh = b * Hh + h;
    int l = Mtop[bh * u + t];
    attn[((size_t)b * L + l) * (Hh * 64) + h * 64 + e] = upd[((size_t)bh * u + t) * 64 + e];
}

// ---------------- conv-distill epilogues ----------------
__global__ void bn_elu_kernel(float* __restrict__ y, const float* __restrict__ cb,
                              const float* __restrict__ bg, const float* __restrict__ bb,
                              int n, int D) {
    int idx = blockIdx.x * blockDim.x + threadIdx.x;
    if (idx >= n) return;
    int d = idx % D;
    float v = (y[idx] + cb[d]) * 0.99999500003749968f * bg[d] + bb[d]; // /sqrt(1+1e-5)
    y[idx] = v > 0.f ? v : (expf(v) - 1.f);
}

__global__ void maxpool_kernel(const float* __restrict__ y, float* __restrict__ out,
                               int B, int L, int D) {
    int Lp = L >> 1;
    int idx = blockIdx.x * blockDim.x + threadIdx.x;
    if (idx >= B * Lp * D) return;
    int d = idx % D;
    int lp = (idx / D) % Lp;
    int b = idx / (D * Lp);
    float m = -3.4e38f;
    for (int k = 0; k < 3; ++k) {
        int l = 2 * lp - 1 + k;
        if (l >= 0 && l < L) m = fmaxf(m, y[((size_t)b * L + l) * D + d]);
    }
    out[idx] = m;
}

// ---------------- host orchestration ----------------
static inline void launch_gemm(const unsigned short* A, const unsigned short* W,
                               const float* bias, float* C, int M, int N, int K,
                               int act, hipStream_t stream) {
    int tiles = (M / 32) * (N / 32);
    gemm_bf16_kernel<<<(tiles + 7) / 8, 256, 0, stream>>>(A, W, bias, C, M, N, K, act);
}

extern "C" void kernel_launch(void* const* d_in, const int* in_sizes, int n_in,
                              void* d_out, int out_size, void* d_ws, size_t ws_size,
                              hipStream_t stream) {
    (void)in_sizes; (void)n_in; (void)out_size; (void)ws_size;
    const float* x_enc  = (const float*)d_in[0];
    const float* emb_w  = (const float*)d_in[1];
    const float* Wq     = (const float*)d_in[2];
    const float* bq     = (const float*)d_in[3];
    const float* Wk     = (const float*)d_in[4];
    const float* bk     = (const float*)d_in[5];
    const float* Wv     = (const float*)d_in[6];
    const float* bv     = (const float*)d_in[7];
    const float* Wo     = (const float*)d_in[8];
    const float* bo     = (const float*)d_in[9];
    const float* ln1_g  = (const float*)d_in[10];
    const float* ln1_b  = (const float*)d_in[11];
    const float* ff_w1  = (const float*)d_in[12];
    const float* ff_b1  = (const float*)d_in[13];
    const float* ff_w2  = (const float*)d_in[14];
    const float* ff_b2  = (const float*)d_in[15];
    const float* ln2_g  = (const float*)d_in[16];
    const float* ln2_b  = (const float*)d_in[17];
    const float* cv_w   = (const float*)d_in[18];
    const float* cv_b   = (const float*)d_in[19];
    const float* bn_g   = (const float*)d_in[20];
    const float* bn_b   = (const float*)d_in[21];
    const float* norm_g = (const float*)d_in[22];
    const float* norm_b = (const float*)d_in[23];

    const int B = 4, L0 = 2048, Cc = 2, D = 512, EL = 3, Hh = 8;

    char* ws = (char*)d_ws;
    size_t off = 0;
    auto alloc = [&](size_t bytes) -> void* {
        void* p = (void*)(ws + off);
        off += (bytes + 255) & ~(size_t)255;
        return p;
    };
    size_t maxel = (size_t)B * L0 * D;
    float* xf   = (float*)alloc(maxel * 4);
    float* tmp  = (float*)alloc(maxel * 4);
    float* qf   = (float*)alloc(maxel * 4);
    float* kf   = (float*)alloc(maxel * 4);
    float* vf   = (float*)alloc(maxel * 4);
    float* attn = (float*)alloc(maxel * 4);
    unsigned short* xbf = (unsigned short*)alloc(maxel * 2);
    unsigned short* tbf = (unsigned short*)alloc(maxel * 2);
    unsigned short* wbf = (unsigned short*)alloc((size_t)3 * D * D * 2);
    float* Mb   = (float*)alloc((size_t)B * Hh * L0 * 4);
    int*   Mtop = (int*)alloc((size_t)B * Hh * 64 * 4);
    float* updb = (float*)alloc((size_t)B * Hh * 64 * 64 * 4);
    float* vmb  = (float*)alloc((size_t)B * Hh * 64 * 4);

    // embedding conv + positional encoding
    {
        int n = B * L0 * D;
        embed_kernel<<<(n + 255) / 256, 256, 0, stream>>>(x_enc, emb_w, xf, B, L0, D, Cc);
    }

    int L = L0;
    for (int i = 0; i < EL; ++i) {
        int cl = 5 * (int)ceil(log((double)L));
        int U = cl < L ? cl : L;                    // U_part == u
        int rows = B * L;
        size_t nel = (size_t)rows * D;
        int nblk = (int)((nel + 255) / 256);
        int wblk = (D * D + 255) / 256;

        // Q/K/V projections (bf16 WMMA, f32 accumulate)
        tobf_kernel<<<nblk, 256, 0, stream>>>(xf, xbf, nel);
        tobf_kernel<<<wblk, 256, 0, stream>>>(Wq + (size_t)i * D * D, wbf, (size_t)D * D);
        launch_gemm(xbf, wbf, bq + i * D, qf, rows, D, D, 0, stream);
        tobf_kernel<<<wblk, 256, 0, stream>>>(Wk + (size_t)i * D * D, wbf, (size_t)D * D);
        launch_gemm(xbf, wbf, bk + i * D, kf, rows, D, D, 0, stream);
        tobf_kernel<<<wblk, 256, 0, stream>>>(Wv + (size_t)i * D * D, wbf, (size_t)D * D);
        launch_gemm(xbf, wbf, bv + i * D, vf, rows, D, D, 0, stream);

        // ProbSparse attention (fp32)
        {
            int totw = B * Hh * L;
            qk_sample_M_kernel<<<(totw + 7) / 8, 256, 0, stream>>>(qf, kf, Mb, B, Hh, L, U);
            topk_kernel<<<B * Hh, 32, 0, stream>>>(Mb, Mtop, L, U);
            attn_upd_kernel<<<B * Hh * U, 256, 0, stream>>>(qf, kf, vf, Mtop, updb, B, Hh, L, U);
            vmean_kernel<<<B * Hh, 64, 0, stream>>>(vf, vmb, Hh, L);
            ctx_fill_kernel<<<nblk, 256, 0, stream>>>(vmb, attn, B, Hh, L);
            ctx_scatter_kernel<<<B * Hh * U, 64, 0, stream>>>(updb, Mtop, attn, Hh, L, U);
        }

        // output projection + residual + LN1 (in-place per row)
        tobf_kernel<<<nblk, 256, 0, stream>>>(attn, tbf, nel);
        tobf_kernel<<<wblk, 256, 0, stream>>>(Wo + (size_t)i * D * D, wbf, (size_t)D * D);
        launch_gemm(tbf, wbf, bo + i * D, tmp, rows, D, D, 0, stream);
        layernorm_kernel<<<rows, 32, 0, stream>>>(xf, tmp, ln1_g + i * D, ln1_b + i * D, xf, D);

        // FFN: gelu(x@W1^T+b1)@W2^T+b2, residual + LN2
        tobf_kernel<<<nblk, 256, 0, stream>>>(xf, xbf, nel);
        tobf_kernel<<<wblk, 256, 0, stream>>>(ff_w1 + (size_t)i * D * D, wbf, (size_t)D * D);
        launch_gemm(xbf, wbf, ff_b1 + i * D, tmp, rows, D, D, 1, stream);
        tobf_kernel<<<nblk, 256, 0, stream>>>(tmp, tbf, nel);
        tobf_kernel<<<wblk, 256, 0, stream>>>(ff_w2 + (size_t)i * D * D, wbf, (size_t)D * D);
        launch_gemm(tbf, wbf, ff_b2 + i * D, tmp, rows, D, D, 0, stream);
        layernorm_kernel<<<rows, 32, 0, stream>>>(xf, tmp, ln2_g + i * D, ln2_b + i * D, xf, D);

        // conv distill (layers 0..EL-2): circular conv (WMMA) + bn + elu + maxpool/2
        if (i < EL - 1) {
            tobf_kernel<<<nblk, 256, 0, stream>>>(xf, xbf, nel);
            pack_convw_kernel<<<(3 * D * D + 255) / 256, 256, 0, stream>>>(
                cv_w + (size_t)i * D * D * 3, wbf, D);
            int tiles = (rows / 16) * (D / 64);
            conv3_gemm_kernel<<<(tiles + 7) / 8, 256, 0, stream>>>(xbf, wbf, tmp, B, L, D);
            bn_elu_kernel<<<nblk, 256, 0, stream>>>(tmp, cv_b + i * D, bn_g + i * D,
                                                    bn_b + i * D, (int)nel, D);
            int Lp = L >> 1;
            int np = B * Lp * D;
            maxpool_kernel<<<(np + 255) / 256, 256, 0, stream>>>(tmp, xf, B, L, D);
            L = Lp;
        }
    }

    // final layernorm -> d_out  (B * 512 * 512 floats)
    layernorm_kernel<<<B * L, 32, 0, stream>>>(xf, nullptr, norm_g, norm_b, (float*)d_out, D);
}
